// EuclideanRiemmanianAtt_987842478683
// MI455X (gfx1250) — matrix operand order repaired and checked by reading, not compiled, not run on hardware
//
#include <hip/hip_runtime.h>
#include <hip/hip_bf16.h>
#include <math.h>

// Problem constants
#define BB 2
#define NN 1024
#define CC 768
#define HH 12
#define DD 64
#define C3 2304   // 3*C
#define H2 24     // 2*H

typedef __attribute__((ext_vector_type(16))) _Float16 v16h;
typedef __attribute__((ext_vector_type(8)))  _Float16 v8h;
typedef __attribute__((ext_vector_type(4)))  _Float16 v4h;
typedef __attribute__((ext_vector_type(8)))  float    v8f;
typedef __attribute__((ext_vector_type(4)))  float    v4f;

__device__ __forceinline__ v16h cat8(v8h a, v8h b) {
  return __builtin_shufflevector(a, b, 0, 1, 2, 3, 4, 5, 6, 7, 8, 9, 10, 11, 12, 13, 14, 15);
}

// ---------------- WMMA fragment loaders (wave32, 16x16x32 f16) ----------------
// A matrix 16x32 (MxK), 16-bit: lane L holds row (L&15); element e maps to
// K = ((e&8)<<1) + (half<<3) + (e&7)  => two contiguous 8-elem runs at
// K = half*8 and K = 16+half*8.  Two 16-byte vector loads.
__device__ __forceinline__ v16h fragA_f16(const _Float16* rowp, int half) {
  v8h lo = *(const v8h*)(rowp + (half << 3));
  v8h hi = *(const v8h*)(rowp + 16 + (half << 3));
  return cat8(lo, hi);
}
// B matrix 32x16 (KxN), 16-bit: lane L holds column (L&15); element e maps to
// K = ((e&8)<<1) + (half<<3) + (e&7)  (same as A => two 16-byte vector loads).
__device__ __forceinline__ v16h fragB_f16(const _Float16* rowp, int half) {
  v8h lo = *(const v8h*)(rowp + (half << 3));
  v8h hi = *(const v8h*)(rowp + 16 + (half << 3));
  return cat8(lo, hi);
}

#define WMMA_F16(a, b, c) \
  __builtin_amdgcn_wmma_f32_16x16x32_f16(false, (a), false, (b), (short)0, (c), false, false)

// ---------------- Pass 0: bulk f32 -> f16 conversion (x, qkv_w, proj_w) ------
__global__ void k_f32_to_f16(const float* __restrict__ src, _Float16* __restrict__ dst,
                             int n4) {
  int idx = blockIdx.x * blockDim.x + threadIdx.x;
  if (idx < n4) {
    v4f v = ((const v4f*)src)[idx];
    v4h h;
#pragma unroll
    for (int e = 0; e < 4; ++e) h[e] = (_Float16)v[e];
    ((v4h*)dst)[idx] = h;
  }
}

// ---------------- Pass 1: QKV GEMM  tmp[m, f] = x[m,:] . qkv_w[f,:] + b[f] ----
// 16x64 tile per wave: 1 A-frag feeds 4 independent WMMAs per k-step.
// grid (2304/64 = 36 n-groups, 128 m-tiles), block 32
__global__ void k_qkv_gemm(const _Float16* __restrict__ x16, const _Float16* __restrict__ w16,
                           const float* __restrict__ bias, float* __restrict__ tmp) {
  const int lane = threadIdx.x & 31, lrow = lane & 15, half = lane >> 4;
  const int n0 = blockIdx.x * 64, m0 = blockIdx.y * 16;
  const _Float16* arow = x16 + (size_t)(m0 + lrow) * CC;
  const _Float16* brow = w16 + (size_t)(n0 + lrow) * CC;
  v8f acc[4];
  v8f zero{};
#pragma unroll
  for (int t = 0; t < 4; ++t) acc[t] = zero;
  for (int kb = 0; kb < CC; kb += 32) {
    __builtin_prefetch(arow + kb + 256, 0, 0);           // global_prefetch_b8
    v16h a = fragA_f16(arow + kb, half);
#pragma unroll
    for (int t = 0; t < 4; ++t)
      acc[t] = WMMA_F16(a, fragB_f16(brow + (size_t)t * 16 * CC + kb, half), acc[t]);
  }
#pragma unroll
  for (int t = 0; t < 4; ++t) {
    int n = n0 + t * 16 + lrow;
    float bi = bias[n];
#pragma unroll
    for (int r = 0; r < 8; ++r)
      tmp[(size_t)(m0 + r + half * 8) * C3 + n] = acc[t][r] + bi;
  }
}

// ---------------- Pass 2: LayerNorm + scatter to q/k f16, v^T f16, norms -----
// one block (256 threads) per token row
__global__ void k_ln_scatter(const float* __restrict__ tmp,
                             const float* __restrict__ g, const float* __restrict__ bta,
                             _Float16* __restrict__ q16, _Float16* __restrict__ k16,
                             _Float16* __restrict__ vT, float* __restrict__ qn2,
                             float* __restrict__ kn2) {
  __shared__ float sh[C3];
  __shared__ float red[256];
  const int tid = threadIdx.x;
  const int row = blockIdx.x;         // b*N + i
  const int b = row >> 10, i = row & 1023;
  const float* src = tmp + (size_t)row * C3;
  float s = 0.f;
#pragma unroll
  for (int t = 0; t < 9; ++t) { float v = src[t * 256 + tid]; sh[t * 256 + tid] = v; s += v; }
  red[tid] = s; __syncthreads();
  for (int st = 128; st > 0; st >>= 1) { if (tid < st) red[tid] += red[tid + st]; __syncthreads(); }
  const float mu = red[0] * (1.0f / C3); __syncthreads();
  float sq = 0.f;
#pragma unroll
  for (int t = 0; t < 9; ++t) { float d = sh[t * 256 + tid] - mu; sq += d * d; }
  red[tid] = sq; __syncthreads();
  for (int st = 128; st > 0; st >>= 1) { if (tid < st) red[tid] += red[tid + st]; __syncthreads(); }
  const float rsig = rsqrtf(red[0] * (1.0f / C3) + 1e-5f); __syncthreads();
#pragma unroll
  for (int t = 0; t < 9; ++t) {
    int f = t * 256 + tid;
    sh[f] = (sh[f] - mu) * rsig * g[f] + bta[f];
  }
  __syncthreads();
#pragma unroll
  for (int t = 0; t < 9; ++t) {
    int f = t * 256 + tid;
    int slot = f / CC, rem = f % CC, h = rem >> 6, d = rem & 63;
    _Float16 y = (_Float16)sh[f];
    size_t hb = (size_t)(b * HH + h);
    if (slot == 0)      q16[(hb * NN + i) * DD + d] = y;
    else if (slot == 1) k16[(hb * NN + i) * DD + d] = y;
    else                vT[(hb * DD + d) * NN + i] = y;     // transposed for AV B-frags
  }
  __syncthreads();
  if (tid < 24) {       // deterministic per-head norm sums (no atomics)
    int slot = tid / HH, h = tid % HH, base = slot * CC + h * 64;
    float acc = 0.f;
#pragma unroll
    for (int d = 0; d < 64; ++d) { float y = sh[base + d]; acc += y * y; }
    float* dst = slot == 0 ? qn2 : kn2;
    dst[(size_t)(b * HH + h) * NN + i] = acc;
  }
}

// ---------------- Pass 3: fold temp*BN*conv into effective affine ------------
__global__ void k_eff(const float* __restrict__ temp, const float* __restrict__ bn_g,
                      const float* __restrict__ bn_b, const float* __restrict__ bn_mean,
                      const float* __restrict__ bn_var, const float* __restrict__ conv_w,
                      const float* __restrict__ conv_b, float* __restrict__ weff,
                      float* __restrict__ beff) {
  __shared__ float alpha[H2], beta[H2];
  int t = threadIdx.x;
  if (t < H2) {
    float inv = rsqrtf(bn_var[t] + 1e-5f);
    alpha[t] = temp[t] * inv * bn_g[t];
    beta[t]  = bn_b[t] - bn_mean[t] * inv * bn_g[t];
  }
  __syncthreads();
  if (t < HH) {
    float be = conv_b[t];
    for (int c = 0; c < H2; ++c) {
      weff[t * H2 + c] = conv_w[t * H2 + c] * alpha[c];
      be += conv_w[t * H2 + c] * beta[c];
    }
    beff[t] = be;
  }
}

// ---------------- Pass 4: fused logits kernel --------------------------------
// For each (b, 16x16 tile): WMMA QK^T for all 12 heads, attn + Riemannian
// branch, fold through effective conv weights into 12 output-head f16 logits.
// grid (64 j-tiles, 64 i-tiles, B), block 32
__global__ void k_logits(const _Float16* __restrict__ q16, const _Float16* __restrict__ k16,
                         const float* __restrict__ qn2, const float* __restrict__ kn2,
                         const float* __restrict__ weff, const float* __restrict__ beff,
                         const float* __restrict__ scale_p, const float* __restrict__ riem_p,
                         _Float16* __restrict__ logits) {
  const int lane = threadIdx.x & 31, lrow = lane & 15, half = lane >> 4;
  const int j0 = blockIdx.x * 16, i0 = blockIdx.y * 16, b = blockIdx.z;
  const float scale = scale_p[0], riem = riem_p[0];
  float acc[HH][8];
#pragma unroll
  for (int o = 0; o < HH; ++o)
#pragma unroll
    for (int r = 0; r < 8; ++r) acc[o][r] = 0.f;

  for (int h = 0; h < HH; ++h) {
    const size_t hb = (size_t)(b * HH + h);
    const _Float16* qrow = q16 + (hb * NN + i0 + lrow) * DD;
    const _Float16* krow = k16 + (hb * NN + j0 + lrow) * DD;
    v8f c{};
    c = WMMA_F16(fragA_f16(qrow, half), fragB_f16(krow, half), c);
    c = WMMA_F16(fragA_f16(qrow + 32, half), fragB_f16(krow + 32, half), c);

    const float kn = kn2[hb * NN + j0 + lrow];      // column norm^2 for this lane
    float fa[8], fr[8];
#pragma unroll
    for (int r = 0; r < 8; ++r) {
      float qn = qn2[hb * NN + i0 + r + half * 8];
      float qk = c[r];
      fa[r] = qk * scale;
      float f2 = qn * qn + kn * kn - 2.0f * qk * qk;
      fr[r] = sqrtf(fmaxf(f2, 1e-12f)) * riem;
    }
#pragma unroll
    for (int o = 0; o < HH; ++o) {
      float w1 = weff[o * H2 + h], w2 = weff[o * H2 + HH + h];
#pragma unroll
      for (int r = 0; r < 8; ++r)
        acc[o][r] = fmaf(w1, fa[r], fmaf(w2, fr[r], acc[o][r]));
    }
  }
#pragma unroll
  for (int o = 0; o < HH; ++o) {
    float bo = beff[o];
#pragma unroll
    for (int r = 0; r < 8; ++r) {
      size_t idx = ((size_t)(b * HH + o) * NN + i0 + r + half * 8) * NN + j0 + lrow;
      logits[idx] = (_Float16)(acc[o][r] + bo);
    }
  }
}

// ---------------- Pass 5: in-place f16 row softmax (f32 internal) ------------
__global__ void k_softmax(_Float16* __restrict__ logits) {
  __shared__ float red[256];
  const int tid = threadIdx.x;
  v4h* p = (v4h*)(logits + (size_t)blockIdx.x * NN);
  v4h hv = p[tid];
  float v[4];
#pragma unroll
  for (int t = 0; t < 4; ++t) v[t] = (float)hv[t];
  float m = fmaxf(fmaxf(v[0], v[1]), fmaxf(v[2], v[3]));
  red[tid] = m; __syncthreads();
  for (int st = 128; st > 0; st >>= 1) { if (tid < st) red[tid] = fmaxf(red[tid], red[tid + st]); __syncthreads(); }
  m = red[0]; __syncthreads();
#pragma unroll
  for (int t = 0; t < 4; ++t) v[t] = expf(v[t] - m);
  red[tid] = v[0] + v[1] + v[2] + v[3]; __syncthreads();
  for (int st = 128; st > 0; st >>= 1) { if (tid < st) red[tid] += red[tid + st]; __syncthreads(); }
  const float inv = 1.0f / red[0];
#pragma unroll
  for (int t = 0; t < 4; ++t) hv[t] = (_Float16)(v[t] * inv);
  p[tid] = hv;
}

// ---------------- Pass 6: AV GEMM, P[N,N] x V[N,D] per output head -----------
// grid (64 i-tiles, H, B), block 32; 4 d-tiles per wave, K loop over N, all f16.
__global__ void k_av(const _Float16* __restrict__ P, const _Float16* __restrict__ vT,
                     _Float16* __restrict__ attc) {
  const int lane = threadIdx.x & 31, lrow = lane & 15, half = lane >> 4;
  const int i0 = blockIdx.x * 16, o = blockIdx.y, b = blockIdx.z;
  const size_t hb = (size_t)(b * HH + o);
  const _Float16* prow = P + (hb * NN + i0 + lrow) * NN;
  v8f acc[4];
  v8f zero{};
#pragma unroll
  for (int t = 0; t < 4; ++t) acc[t] = zero;
  for (int kb = 0; kb < NN; kb += 32) {
    __builtin_prefetch(prow + kb + 256, 0, 0);
    v16h a = fragA_f16(prow + kb, half);
#pragma unroll
    for (int t = 0; t < 4; ++t) {
      const _Float16* vrow = vT + (hb * DD + t * 16 + lrow) * NN + kb;
      acc[t] = WMMA_F16(a, fragB_f16(vrow, half), acc[t]);
    }
  }
#pragma unroll
  for (int t = 0; t < 4; ++t)
#pragma unroll
    for (int r = 0; r < 8; ++r) {
      int m = i0 + r + half * 8, col = o * DD + t * 16 + lrow;
      attc[(size_t)(b * NN + m) * CC + col] = (_Float16)acc[t][r];
    }
}

// ---------------- Pass 7: output projection GEMM + bias ----------------------
// 16x64 tile per wave; grid (768/64 = 12 n-groups, 128 m-tiles), block 32
__global__ void k_proj(const _Float16* __restrict__ attc, const _Float16* __restrict__ w16,
                       const float* __restrict__ bias, float* __restrict__ out) {
  const int lane = threadIdx.x & 31, lrow = lane & 15, half = lane >> 4;
  const int n0 = blockIdx.x * 64, m0 = blockIdx.y * 16;
  const _Float16* arow = attc + (size_t)(m0 + lrow) * CC;
  const _Float16* brow = w16 + (size_t)(n0 + lrow) * CC;
  v8f acc[4];
  v8f zero{};
#pragma unroll
  for (int t = 0; t < 4; ++t) acc[t] = zero;
  for (int kb = 0; kb < CC; kb += 32) {
    __builtin_prefetch(arow + kb + 256, 0, 0);
    v16h a = fragA_f16(arow + kb, half);
#pragma unroll
    for (int t = 0; t < 4; ++t)
      acc[t] = WMMA_F16(a, fragB_f16(brow + (size_t)t * 16 * CC + kb, half), acc[t]);
  }
#pragma unroll
  for (int t = 0; t < 4; ++t) {
    int n = n0 + t * 16 + lrow;
    float bi = bias[n];
#pragma unroll
    for (int r = 0; r < 8; ++r)
      out[(size_t)(m0 + r + half * 8) * CC + n] = acc[t][r] + bi;
  }
}

// ---------------- Workspace layout (bytes, 256-aligned) ----------------------
#define TMP_OFF    0u                      // 2048*2304 f32      = 18,874,368
#define Q16_OFF    18874368u               // 2*12*1024*64 f16   =  3,145,728
#define K16_OFF    22020096u
#define VT16_OFF   25165824u
#define QN2_OFF    28311552u               // 24576 f32
#define KN2_OFF    28409856u
#define WEFF_OFF   28508160u               // 288 f32 (padded)
#define BEFF_OFF   28509440u               // 12 f32 (padded)
#define LG16_OFF   28509696u               // 2*12*1024*1024 f16 = 50,331,648
#define ATTC_OFF   78841344u               // 2048*768 f16       =  3,145,728
#define X16_OFF    81987072u               // 2048*768 f16       =  3,145,728
#define W16_OFF    85132800u               // 2304*768 f16       =  3,538,944
#define PW16_OFF   88671744u               // 768*768 f16        =  1,179,648
// total 89,851,392 bytes (~85.7 MB)

extern "C" void kernel_launch(void* const* d_in, const int* in_sizes, int n_in,
                              void* d_out, int out_size, void* d_ws, size_t ws_size,
                              hipStream_t stream) {
  const float* x       = (const float*)d_in[0];
  const float* qkv_w   = (const float*)d_in[1];
  const float* qkv_b   = (const float*)d_in[2];
  const float* ln_g    = (const float*)d_in[3];
  const float* ln_b    = (const float*)d_in[4];
  const float* scale   = (const float*)d_in[5];
  const float* riem    = (const float*)d_in[6];
  const float* temp    = (const float*)d_in[7];
  const float* bn_g    = (const float*)d_in[8];
  const float* bn_b    = (const float*)d_in[9];
  const float* bn_mean = (const float*)d_in[10];
  const float* bn_var  = (const float*)d_in[11];
  const float* conv_w  = (const float*)d_in[12];
  const float* conv_b  = (const float*)d_in[13];
  const float* proj_w  = (const float*)d_in[14];
  const float* proj_b  = (const float*)d_in[15];
  float* out = (float*)d_out;

  char* ws = (char*)d_ws;
  float*     tmp  = (float*)(ws + TMP_OFF);
  _Float16*  q16  = (_Float16*)(ws + Q16_OFF);
  _Float16*  k16  = (_Float16*)(ws + K16_OFF);
  _Float16*  vT   = (_Float16*)(ws + VT16_OFF);
  float*     qn2  = (float*)(ws + QN2_OFF);
  float*     kn2  = (float*)(ws + KN2_OFF);
  float*     weff = (float*)(ws + WEFF_OFF);
  float*     beff = (float*)(ws + BEFF_OFF);
  _Float16*  lg   = (_Float16*)(ws + LG16_OFF);
  _Float16*  attc = (_Float16*)(ws + ATTC_OFF);
  _Float16*  x16  = (_Float16*)(ws + X16_OFF);
  _Float16*  w16  = (_Float16*)(ws + W16_OFF);
  _Float16*  pw16 = (_Float16*)(ws + PW16_OFF);

  // up-front f32 -> f16 conversions (removes cvt chains from GEMM hot loops)
  k_f32_to_f16<<<(2048 * 768 / 4 + 255) / 256, 256, 0, stream>>>(x, x16, 2048 * 768 / 4);
  k_f32_to_f16<<<(2304 * 768 / 4 + 255) / 256, 256, 0, stream>>>(qkv_w, w16, 2304 * 768 / 4);
  k_f32_to_f16<<<(768 * 768 / 4 + 255) / 256, 256, 0, stream>>>(proj_w, pw16, 768 * 768 / 4);

  k_qkv_gemm <<<dim3(C3 / 64, (BB * NN) / 16), 32, 0, stream>>>(x16, w16, qkv_b, tmp);
  k_ln_scatter<<<BB * NN, 256, 0, stream>>>(tmp, ln_g, ln_b, q16, k16, vT, qn2, kn2);
  k_eff      <<<1, 32, 0, stream>>>(temp, bn_g, bn_b, bn_mean, bn_var, conv_w, conv_b, weff, beff);
  k_logits   <<<dim3(NN / 16, NN / 16, BB), 32, 0, stream>>>(q16, k16, qn2, kn2, weff, beff,
                                                             scale, riem, lg);
  k_softmax  <<<BB * HH * NN, 256, 0, stream>>>(lg);
  k_av       <<<dim3(NN / 16, HH, BB), 32, 0, stream>>>(lg, vT, attc);
  k_proj     <<<dim3(CC / 16 / 4, (BB * NN) / 16), 32, 0, stream>>>(attc, pw16, proj_b, out);
}
